// SparseAttention_5093831213622
// MI455X (gfx1250) — compile-verified, matrix operand order checked by reading, not run on hardware
//
#include <hip/hip_runtime.h>
#include <hip/hip_bf16.h>

// Problem dims (fixed by the reference)
#define BB   2
#define SS   2048
#define HH   768
#define NHH  12
#define HDD  64

typedef __attribute__((ext_vector_type(16))) __bf16 v16bf;
typedef __attribute__((ext_vector_type(8)))  __bf16 v8bf;
typedef __attribute__((ext_vector_type(8)))  float  v8f;
typedef __attribute__((ext_vector_type(4)))  float  v4f;

// ---------------------------------------------------------------------------
// Pack a 16x32 bf16 operand tile (MxK, row-major source, leading dim `ld`)
// into the CDNA5 WMMA 16-bit A/B register layout:
//   lanes 0-15  : row = lane,    K = {k0+0..7,  k0+16..23}
//   lanes 16-31 : row = lane-16, K = {k0+8..15, k0+24..31}
// For an NT GEMM both the A (Q rows) and B (K rows = columns of K^T) operands
// use this identical gather.
// ---------------------------------------------------------------------------
__device__ __forceinline__ v16bf pack_from_bf16(const __bf16* __restrict__ base,
                                                int ld, int row0, int k0, int lane) {
    const int m    = lane & 15;
    const int koff = (lane < 16) ? 0 : 8;
    const __bf16* rp = base + (size_t)(row0 + m) * ld + k0 + koff;
    v8bf lo = *(const v8bf*)(rp);        // 16B global_load_b128
    v8bf hi = *(const v8bf*)(rp + 16);   // 16B global_load_b128
    v16bf r;
#pragma unroll
    for (int i = 0; i < 8; ++i) { r[i] = lo[i]; r[i + 8] = hi[i]; }
    return r;
}

// Same tile gather, but from an f32 source with inline f32->bf16 conversion
// (used by the projection kernel for activations and weights).
__device__ __forceinline__ v16bf pack_from_f32(const float* __restrict__ base,
                                               int ld, int row0, int k0, int lane) {
    const int m    = lane & 15;
    const int koff = (lane < 16) ? 0 : 8;
    const float* rp = base + (size_t)(row0 + m) * ld + k0 + koff;
    v4f a0 = *(const v4f*)(rp + 0);
    v4f a1 = *(const v4f*)(rp + 4);
    v4f b0 = *(const v4f*)(rp + 16);
    v4f b1 = *(const v4f*)(rp + 20);
    v16bf r;
#pragma unroll
    for (int i = 0; i < 4; ++i) {
        r[i]      = (__bf16)a0[i];
        r[i + 4]  = (__bf16)a1[i];
        r[i + 8]  = (__bf16)b0[i];
        r[i + 12] = (__bf16)b1[i];
    }
    return r;
}

// ---------------------------------------------------------------------------
// Kernel 1: fused Q/K projection.  grid = (M/16, (H/16)/4, 2), block = 128.
//   z==0 -> Q = X @ Wq^T + bq ;  z==1 -> K = X @ Wk^T + bk
// Each wave computes one 16x16 output tile over the K=768 reduction.
// TWO independent accumulators (even/odd K-steps) break the WMMA->WMMA D->C
// dependence chain so consecutive v_wmma issue without hazard v_nops; the
// partial sums are combined in the epilogue.
// ---------------------------------------------------------------------------
__global__ void __launch_bounds__(128)
proj_qk_bf16(const float* __restrict__ X,    // [B*S, H] f32
             const float* __restrict__ Wq,   // [H, H]  f32 (row n = output feature)
             const float* __restrict__ bq,   // [H]
             const float* __restrict__ Wk,   // [H, H]
             const float* __restrict__ bk,   // [H]
             __bf16* __restrict__ Qo,        // [B*S, H] bf16
             __bf16* __restrict__ Ko) {      // [B*S, H] bf16
    const int lane = threadIdx.x & 31;
    const int wave = threadIdx.x >> 5;
    const int m0 = blockIdx.x * 16;
    const int n0 = (blockIdx.y * 4 + wave) * 16;

    const float* W    = (blockIdx.z == 0) ? Wq : Wk;
    const float* bias = (blockIdx.z == 0) ? bq : bk;
    __bf16*      Out  = (blockIdx.z == 0) ? Qo : Ko;

    v8f acc0 = {};
    v8f acc1 = {};
#pragma unroll 4
    for (int k0 = 0; k0 < HH; k0 += 64) {   // two independent 32-wide K steps
        v16bf a0 = pack_from_f32(X, HH, m0, k0,      lane);
        v16bf b0 = pack_from_f32(W, HH, n0, k0,      lane);
        acc0 = __builtin_amdgcn_wmma_f32_16x16x32_bf16(false, a0, false, b0,
                                                       (short)0, acc0, false, false);
        v16bf a1 = pack_from_f32(X, HH, m0, k0 + 32, lane);
        v16bf b1 = pack_from_f32(W, HH, n0, k0 + 32, lane);
        acc1 = __builtin_amdgcn_wmma_f32_16x16x32_bf16(false, a1, false, b1,
                                                       (short)0, acc1, false, false);
    }

    // C/D layout: lane n (0-15) holds column n0+n; vgpr r -> row m0+r (+8 for hi lanes)
    const int n    = n0 + (lane & 15);
    const int moff = (lane < 16) ? 0 : 8;
    const float bn = bias[n];
#pragma unroll
    for (int r = 0; r < 8; ++r) {
        Out[(size_t)(m0 + moff + r) * HH + n] = (__bf16)(acc0[r] + acc1[r] + bn);
    }
}

// ---------------------------------------------------------------------------
// Kernel 2: per-(batch,head) scores = relu(Q K^T / 8 + mask[q]),
// written directly in [B, S, NH, S] layout with non-temporal stores
// (403 MB stream-out > 192 MB L2 -> do not cache the output).
// grid = ((S/16)/4, S/256, B*NH), block = 128.
// Each wave: one 16-row strip of 256 columns; A-tile reused across 16 column
// tiles; K=64 -> 2 independent WMMAs per tile (summed in the epilogue fma).
// All 4 waves of a block share the same column strip -> B tiles hit in L1.
// ---------------------------------------------------------------------------
__global__ void __launch_bounds__(128)
scores_relu_kernel(const __bf16* __restrict__ Q,    // [B*S, H] bf16
                   const __bf16* __restrict__ K,    // [B*S, H] bf16
                   const float* __restrict__ mask,  // [B, S]
                   float* __restrict__ out) {       // [B, S, NH, S] f32
    const int lane = threadIdx.x & 31;
    const int wave = threadIdx.x >> 5;
    const int p = blockIdx.z;            // 0..B*NH-1
    const int b = p / NHH;
    const int h = p % NHH;
    const int m0     = (blockIdx.x * 4 + wave) * 16;  // query-row tile
    const int strip0 = blockIdx.y * 256;              // key-column strip base

    const __bf16* Qb = Q + (size_t)b * SS * HH + (size_t)h * HDD;
    const __bf16* Kb = K + (size_t)b * SS * HH + (size_t)h * HDD;

    // A operand: Q rows m0..m0+15, full head dim (K=64), loaded once.
    v16bf a0 = pack_from_bf16(Qb, HH, m0, 0,  lane);
    v16bf a1 = pack_from_bf16(Qb, HH, m0, 32, lane);

    // Per-row additive mask (mask is over the QUERY axis in the reference).
    const int moff = (lane < 16) ? 0 : 8;
    float mrow[8];
#pragma unroll
    for (int r = 0; r < 8; ++r)
        mrow[r] = mask[(size_t)b * SS + m0 + moff + r];

    const int ncol = lane & 15;
    const size_t mstride = (size_t)NHH * SS;  // floats between consecutive query rows
    float* obase = out + (((size_t)b * SS + (m0 + moff)) * NHH + h) * SS + strip0 + ncol;

#pragma unroll 2
    for (int nt = 0; nt < 16; ++nt) {
        const int n0 = strip0 + nt * 16;
        // Prefetch next column tile's K rows while this tile computes.
        if (nt + 1 < 16)
            __builtin_prefetch(Kb + (size_t)(n0 + 16 + ncol) * HH, 0, 0);

        v16bf b0 = pack_from_bf16(Kb, HH, n0, 0,  lane);
        v16bf b1 = pack_from_bf16(Kb, HH, n0, 32, lane);
        v8f acc0 = {};
        v8f acc1 = {};
        acc0 = __builtin_amdgcn_wmma_f32_16x16x32_bf16(false, a0, false, b0,
                                                       (short)0, acc0, false, false);
        acc1 = __builtin_amdgcn_wmma_f32_16x16x32_bf16(false, a1, false, b1,
                                                       (short)0, acc1, false, false);
#pragma unroll
        for (int r = 0; r < 8; ++r) {
            float v = fmaf(acc0[r] + acc1[r], 0.125f, mrow[r]); // 1/sqrt(64) + mask
            v = (v > 0.0f) ? v : 0.0f;                          // relu
            __builtin_nontemporal_store(v, obase + (size_t)r * mstride + nt * 16);
        }
    }
}

// ---------------------------------------------------------------------------
// Host-side launch.  Inputs (setup_inputs order):
//  0 hidden_states f32 [B,S,H], 1 attention_mask f32 [B,S], 2 input_ids (unused),
//  3 ngram (unused), 4 Wq, 5 bq, 6 Wk, 7 bk.  Output f32 [B,S,NH,S].
// Workspace: Q bf16 (6.29 MB) + K bf16 (6.29 MB).
// ---------------------------------------------------------------------------
extern "C" void kernel_launch(void* const* d_in, const int* in_sizes, int n_in,
                              void* d_out, int out_size, void* d_ws, size_t ws_size,
                              hipStream_t stream) {
    const float* X    = (const float*)d_in[0];
    const float* mask = (const float*)d_in[1];
    const float* Wq   = (const float*)d_in[4];
    const float* bq   = (const float*)d_in[5];
    const float* Wk   = (const float*)d_in[6];
    const float* bk   = (const float*)d_in[7];
    float* out = (float*)d_out;

    __bf16* Qo = (__bf16*)d_ws;
    __bf16* Ko = Qo + (size_t)BB * SS * HH;

    dim3 gridA(BB * SS / 16, (HH / 16) / 4, 2);      // (256, 12, 2), 4 waves/block
    proj_qk_bf16<<<gridA, 128, 0, stream>>>(X, Wq, bq, Wk, bk, Qo, Ko);

    dim3 gridB((SS / 16) / 4, SS / 256, BB * NHH);   // (32, 8, 24), 4 waves/block
    scores_relu_kernel<<<gridB, 128, 0, stream>>>(Qo, Ko, mask, out);
}
